// ForecastNetSimple_39659728011617
// MI455X (gfx1250) — matrix operand order repaired and checked by reading, not compiled, not run on hardware
//
#include <hip/hip_runtime.h>
#include <hip/hip_bf16.h>
#include <math.h>

// ---------------------------------------------------------------------------
// ForecastNet seq2seq LSTM for MI455X (gfx1250, wave32, WMMA)
//
// Per-step recurrent GEMM: (128 x 512) x (512 x 2048), f16 in / f32 acc via
// v_wmma_f32_16x16x32_f16. One wave owns one 16x16 h-tile and computes all
// FOUR gate tiles (i/f/g/o) reusing a single A fragment per k-step, so the
// LSTM cell update is pure in-wave register math (no LDS, no barriers).
// Weights (2MB f16) + state (<1MB) stay resident in the 192MB L2; the
// sequential recurrence runs as one launch per timestep (graph-replayed).
// k-loop unrolled x4: at ~1 wave/SIMD occupancy, load latency must be hidden
// by in-wave ILP, so we give the scheduler a 16-WMMA window of independent
// global_load_b128 streams.
// ---------------------------------------------------------------------------

typedef __attribute__((ext_vector_type(16))) _Float16 v16h;
typedef __attribute__((ext_vector_type(8)))  _Float16 v8h;
typedef __attribute__((ext_vector_type(8)))  float    v8f;

static constexpr int kB     = 128;   // batch
static constexpr int kT     = 512;   // encoder sequence length
static constexpr int kI     = 8;     // input features
static constexpr int kH     = 512;   // hidden
static constexpr int kFourH = 4 * kH;
static constexpr int kW     = 96;    // decoder horizon

__device__ __forceinline__ float sigmoidf_(float x) {
  return 1.0f / (1.0f + __expf(-x));
}

// --------------------------- prep ------------------------------------------
// Convert Whh matrices to f16, fuse biases, zero the parity-0 state buffers.
// Grid: (kFourH*kH)/256 blocks x 256 threads => i covers exactly kFourH*kH.
__global__ void prep_kernel(const float* __restrict__ WhhE,
                            const float* __restrict__ WhhD,
                            const float* __restrict__ bihE,
                            const float* __restrict__ bhhE,
                            const float* __restrict__ bihD,
                            const float* __restrict__ bhhD,
                            _Float16* __restrict__ whhE16,
                            _Float16* __restrict__ whhD16,
                            float* __restrict__ bE,
                            float* __restrict__ bD,
                            _Float16* __restrict__ h0,
                            float* __restrict__ c0) {
  int i = blockIdx.x * blockDim.x + threadIdx.x;
  whhE16[i] = (_Float16)WhhE[i];
  whhD16[i] = (_Float16)WhhD[i];
  if (i < kFourH) {
    bE[i] = bihE[i] + bhhE[i];
    bD[i] = bihD[i] + bhhD[i];
  }
  if (i < kB * kH) {
    h0[i] = (_Float16)0.0f;
    c0[i] = 0.0f;
  }
}

// --------------------------- LSTM step -------------------------------------
// One timestep. Grid: (32 n-tiles, 8 m-tiles), one wave (32 threads) per
// block. The wave computes four 16x16 gate tiles with a shared A fragment:
// per 32-wide k-step => 1 A load + 4 B loads feeding 4 WMMAs.
__global__ __launch_bounds__(32) void lstm_step_kernel(
    const _Float16* __restrict__ hprev,   // [kB][kH] f16 (parity p)
    const float*    __restrict__ cprev,   // [kB][kH] f32 (parity p)
    const _Float16* __restrict__ Whh16,   // [kFourH][kH] f16
    const float*    __restrict__ Wih,     // [kFourH][kI] f32
    const float*    __restrict__ bias,    // [kFourH] f32 (bih+bhh)
    const float*    __restrict__ x,       // x[m*xstride + q], q<kI
    int xstride,
    _Float16* __restrict__ hnew,          // [kB][kH] f16 (parity p^1)
    float*    __restrict__ cnew) {        // [kB][kH] f32 (parity p^1)
  const int lane = threadIdx.x & 31;
  const int m0   = blockIdx.y * 16;       // batch-tile base
  const int n0   = blockIdx.x * 16;       // hidden-tile base

  // A fragment (16x32 f16): lane<16 -> M=lane, K = k0+0..15 contiguous;
  // lane>=16 -> M=lane-16, K = k0+16..31 (matches B).
  const int arow = m0 + (lane & 15);
  const int asel = (lane >> 4) << 4;                 // +0 / +16
  const _Float16* aptr = hprev + (size_t)arow * kH + asel;

  // B fragments (32x16 f16): lane<16 -> N=lane, K=k0..k0+15 contiguous;
  // lane>=16 -> N=lane-16, K=k0+16..k0+31.  B(k,n)=Whh[n][k] -> row reads.
  // Gate g's rows live at offset g*kH rows => g*kH*kH elements.
  const int nlane = lane & 15;
  const int bsel  = (lane >> 4) << 4;                // +0 / +16
  const _Float16* bptr = Whh16 + (size_t)(n0 + nlane) * kH + bsel;
  const size_t gstr = (size_t)kH * kH;               // gate stride (elements)

  v8f acc0 = {}, acc1 = {}, acc2 = {}, acc3 = {};
#pragma unroll 4
  for (int k0 = 0; k0 < kH; k0 += 32) {
    union { v16h v; v8h h[2]; } a;
    a.v = *(const v16h*)(aptr + k0);
    v16h b0 = *(const v16h*)(bptr + k0);
    v16h b1 = *(const v16h*)(bptr + gstr + k0);
    v16h b2 = *(const v16h*)(bptr + 2 * gstr + k0);
    v16h b3 = *(const v16h*)(bptr + 3 * gstr + k0);
    acc0 = __builtin_amdgcn_wmma_f32_16x16x32_f16(false, a.v, false, b0,
                                                  (short)0, acc0, false, false);
    acc1 = __builtin_amdgcn_wmma_f32_16x16x32_f16(false, a.v, false, b1,
                                                  (short)0, acc1, false, false);
    acc2 = __builtin_amdgcn_wmma_f32_16x16x32_f16(false, a.v, false, b2,
                                                  (short)0, acc2, false, false);
    acc3 = __builtin_amdgcn_wmma_f32_16x16x32_f16(false, a.v, false, b3,
                                                  (short)0, acc3, false, false);
  }

  // Epilogue, all in-wave. D layout: VGPR r -> M = r (+8 for lanes>=16),
  // N = lane&15. Each lane owns one hidden column jn for 8 batch rows.
  const int jn = n0 + nlane;
  const float b_i = bias[jn];
  const float b_f = bias[kH + jn];
  const float b_g = bias[2 * kH + jn];
  const float b_o = bias[3 * kH + jn];

  float wi[kI], wf[kI], wg[kI], wo[kI];
#pragma unroll
  for (int q = 0; q < kI; ++q) {
    wi[q] = Wih[jn * kI + q];
    wf[q] = Wih[(kH + jn) * kI + q];
    wg[q] = Wih[(2 * kH + jn) * kI + q];
    wo[q] = Wih[(3 * kH + jn) * kI + q];
  }

  const int mbase = m0 + ((lane >> 4) << 3);
#pragma unroll
  for (int r = 0; r < 8; ++r) {
    const float* xr = x + (size_t)(mbase + r) * xstride;
    float si = b_i, sf = b_f, sg = b_g, so = b_o;
#pragma unroll
    for (int q = 0; q < kI; ++q) {
      const float xq = xr[q];
      si += xq * wi[q];
      sf += xq * wf[q];
      sg += xq * wg[q];
      so += xq * wo[q];
    }
    const float ig = sigmoidf_(acc0[r] + si);
    const float fg = sigmoidf_(acc1[r] + sf);
    const float gg = tanhf(acc2[r] + sg);
    const float og = sigmoidf_(acc3[r] + so);
    const int idx = (mbase + r) * kH + jn;
    const float cn = fg * cprev[idx] + ig * gg;
    const float hn = og * tanhf(cn);
    cnew[idx] = cn;
    hnew[idx] = (_Float16)hn;
  }
}

// --------------------------- decoder input ---------------------------------
// xp[m][q] = x[m]*Wp[q] + bp[q].  128 threads, one per batch row.
__global__ void dec_input_kernel(const float* __restrict__ xsrc, int xstride,
                                 const float* __restrict__ Wp,
                                 const float* __restrict__ bp,
                                 float* __restrict__ xp) {
  const int m = threadIdx.x;
  const float xm = xsrc[(size_t)m * xstride];
#pragma unroll
  for (int q = 0; q < kI; ++q) xp[m * kI + q] = xm * Wp[q] + bp[q];
}

// --------------------------- decoder prediction -----------------------------
// pred[m] = sigmoid(h[m] . Wfc + bfc); write to output[:, d] and feedback buf.
__global__ void dec_pred_kernel(const _Float16* __restrict__ h,
                                const float* __restrict__ Wfc,
                                const float* __restrict__ bfc,
                                float* __restrict__ predbuf,
                                float* __restrict__ out, int d) {
  const int m = threadIdx.x;               // 128 threads
  float s = bfc[0];
  for (int k = 0; k < kH; ++k) s += (float)h[m * kH + k] * Wfc[k];
  const float p = sigmoidf_(s);
  predbuf[m] = p;
  out[m * kW + d] = p;
}

// --------------------------- host driver ------------------------------------
extern "C" void kernel_launch(void* const* d_in, const int* in_sizes, int n_in,
                              void* d_out, int out_size, void* d_ws,
                              size_t ws_size, hipStream_t stream) {
  (void)in_sizes; (void)n_in; (void)out_size; (void)ws_size;

  const float* src  = (const float*)d_in[0];   // [B][T][I]
  const float* WihE = (const float*)d_in[1];   // [4H][I]
  const float* WhhE = (const float*)d_in[2];   // [4H][H]
  const float* bihE = (const float*)d_in[3];
  const float* bhhE = (const float*)d_in[4];
  const float* WihD = (const float*)d_in[5];
  const float* WhhD = (const float*)d_in[6];
  const float* bihD = (const float*)d_in[7];
  const float* bhhD = (const float*)d_in[8];
  const float* Wp   = (const float*)d_in[9];   // [I][1]
  const float* bp   = (const float*)d_in[10];  // [I]
  const float* Wfc  = (const float*)d_in[11];  // [1][H]
  const float* bfc  = (const float*)d_in[12];  // [1]
  float* out = (float*)d_out;                  // [B][W][1]

  // Workspace carve-out (256B aligned), ~4.8 MB total.
  char* ws = (char*)d_ws;
  size_t off = 0;
  auto carve = [&](size_t bytes) -> void* {
    void* p = ws + off;
    off = (off + bytes + 255) & ~(size_t)255;
    return p;
  };
  _Float16* whhE16 = (_Float16*)carve((size_t)kFourH * kH * sizeof(_Float16));
  _Float16* whhD16 = (_Float16*)carve((size_t)kFourH * kH * sizeof(_Float16));
  float*    bE     = (float*)carve(kFourH * sizeof(float));
  float*    bD     = (float*)carve(kFourH * sizeof(float));
  _Float16* h16[2];
  h16[0] = (_Float16*)carve((size_t)kB * kH * sizeof(_Float16));
  h16[1] = (_Float16*)carve((size_t)kB * kH * sizeof(_Float16));
  float* c32[2];
  c32[0] = (float*)carve((size_t)kB * kH * sizeof(float));
  c32[1] = (float*)carve((size_t)kB * kH * sizeof(float));
  float* xp   = (float*)carve(kB * kI * sizeof(float));
  float* pred = (float*)carve(kB * sizeof(float));

  // Prep: f16 weight copies, fused biases, zeroed parity-0 state.
  prep_kernel<<<(kFourH * kH) / 256, 256, 0, stream>>>(
      WhhE, WhhD, bihE, bhhE, bihD, bhhD, whhE16, whhD16, bE, bD,
      h16[0], c32[0]);

  const dim3 gemmGrid(kH / 16, kB / 16);  // 32 x 8 single-wave workgroups
  int par = 0;

  // Encoder: 512 dependent steps. x_t(m,q) = src[m*T*I + t*I + q].
  for (int t = 0; t < kT; ++t) {
    lstm_step_kernel<<<gemmGrid, 32, 0, stream>>>(
        h16[par], c32[par], whhE16, WihE, bE,
        src + (size_t)t * kI, kT * kI,
        h16[par ^ 1], c32[par ^ 1]);
    par ^= 1;
  }

  // Decoder: 96 steps with scalar feedback.
  for (int d = 0; d < kW; ++d) {
    if (d == 0) {
      // x0 = src[:, T-1, 0]
      dec_input_kernel<<<1, kB, 0, stream>>>(
          src + (size_t)(kT - 1) * kI, kT * kI, Wp, bp, xp);
    } else {
      dec_input_kernel<<<1, kB, 0, stream>>>(pred, 1, Wp, bp, xp);
    }
    lstm_step_kernel<<<gemmGrid, 32, 0, stream>>>(
        h16[par], c32[par], whhD16, WihD, bD, xp, kI,
        h16[par ^ 1], c32[par ^ 1]);
    par ^= 1;
    dec_pred_kernel<<<1, kB, 0, stream>>>(h16[par], Wfc, bfc, pred, out, d);
  }
}
